// RvT_14628658610870
// MI455X (gfx1250) — compile-verified, matrix operand order checked by reading, not run on hardware
//
#include <hip/hip_runtime.h>
#include <hip/hip_bf16.h>

typedef __attribute__((ext_vector_type(16))) _Float16 v16h;
typedef __attribute__((ext_vector_type(8)))  _Float16 v8h;
typedef __attribute__((ext_vector_type(8)))  float    v8f;

#define DEV static __device__ __forceinline__

static constexpr int NB = 16, CIN = 128, COUT = 256, G = 8, CG = 32;
static constexpr int H = 28, HW = 784, KD = 676, VROW = 680; // VROW: padded v row (16B mult)

// Load a WMMA 16-bit operand half-pair: VGPR0-3 from p[0..7], VGPR4-7 from p[16..23].
// Caller folds in (lane>=16 ? +8 : 0). Requires 16B alignment of p.
DEV v16h ldpair(const _Float16* p) {
  v8h a = *(const v8h*)p;
  v8h b = *(const v8h*)(p + 16);
  v16h r;
#pragma unroll
  for (int i = 0; i < 8; ++i) { r[i] = a[i]; r[i + 8] = b[i]; }
  return r;
}

DEV v8f wmma16(v16h a, v16h b, v8f c) {
  return __builtin_amdgcn_wmma_f32_16x16x32_f16(false, a, false, b, (short)0, c,
                                                false, false);
}

DEV float sigm(float x) { return 1.0f / (1.0f + __expf(-x)); }
DEV float ftanh(float x) {
  x = fminf(15.0f, fmaxf(-15.0f, x));
  float t = __expf(2.0f * x);
  return (t - 1.0f) / (t + 1.0f);
}

// ---------------- prep kernels ----------------
__global__ void k_zero16(_Float16* p, int n) {
  int i = blockIdx.x * 256 + threadIdx.x;
  if (i < n) p[i] = (_Float16)0.0f;
}

__global__ void k_cvt(const float* s, _Float16* d, int n) {
  int i = blockIdx.x * 256 + threadIdx.x;
  if (i < n) d[i] = (_Float16)s[i];
}

// w[COUT][2*CG][3][3] -> f16 [g][tap][oc(32)][ic(32)], x-half only (ic 0..31)
__global__ void k_cvt_wconv(const float* s, _Float16* d) {
  int i = blockIdx.x * 256 + threadIdx.x;
  if (i >= G * 9 * CG * CG) return;
  int ic = i & 31, oc = (i >> 5) & 31, t = (i / 1024) % 9, g = i / 9216;
  d[i] = (_Float16)s[(((size_t)(g * 32 + oc) * 64 + ic) * 9) + t];
}

// x_in [N][CIN][HW] f32 -> [n][pos][ch] f16
__global__ void k_xt(const float* x, _Float16* d) {
  int i = blockIdx.x * 256 + threadIdx.x;
  if (i >= NB * HW * CIN) return;
  int c = i & 127, p = (i >> 7) % HW, n = i / (HW * CIN);
  d[i] = (_Float16)x[((size_t)n * CIN + c) * HW + p];
}

// ---------------- proj: XH = Wx @ x_in (1x1 conv) ----------------
__global__ void __launch_bounds__(32) k_proj(const _Float16* __restrict__ XT,
                                             const _Float16* __restrict__ WX,
                                             _Float16* __restrict__ XH) {
  int id = blockIdx.x;
  int pt = id % 49, mt = (id / 49) & 15, n = id / (49 * 16);
  int l = threadIdx.x, hs = l >> 4, ln = l & 15;
  int p = pt * 16 + ln;
  v8f acc = {0, 0, 0, 0, 0, 0, 0, 0};
#pragma unroll
  for (int kc = 0; kc < 4; ++kc) {
    v16h B = ldpair(XT + ((size_t)n * HW + p) * CIN + kc * 32 + hs * 8);
    v16h A = ldpair(WX + (size_t)(mt * 16 + ln) * CIN + kc * 32 + hs * 8);
    acc = wmma16(A, B, acc);
  }
#pragma unroll
  for (int j = 0; j < 8; ++j) {
    int oc = mt * 16 + j + 8 * hs;
    XH[((size_t)n * HW + p) * COUT + oc] = (_Float16)acc[j];
  }
}

// ---------------- grouped 3x3 conv (implicit GEMM) ----------------
// mode 0: q SAME -> [n][g][pos][ch]; 1: k VALID -> [n][g][key][ch]; 2: v VALID -> [n][g][ch][VROW]
__global__ void __launch_bounds__(32) k_conv3(const _Float16* __restrict__ XH,
                                              const _Float16* __restrict__ Wc,
                                              _Float16* __restrict__ Out,
                                              const _Float16* __restrict__ zp, int mode) {
  int PT = (mode == 0) ? 49 : 43;
  int id = blockIdx.x;
  int pt = id % PT; id /= PT;
  int mt = id & 1; id >>= 1;
  int g = id & 7, n = id >> 3;
  int l = threadIdx.x, hs = l >> 4, ln = l & 15;
  int p = pt * 16 + ln;
  int y, x;
  if (mode == 0) { y = p / 28; x = p % 28; }
  else { int pc = p < 676 ? p : 675; y = pc / 26; x = pc % 26; }
  v8f acc = {0, 0, 0, 0, 0, 0, 0, 0};
#pragma unroll
  for (int t = 0; t < 9; ++t) {
    int dy = t / 3, dx = t % 3;
    bool valid = true;
    int iy, ix;
    if (mode == 0) { iy = y + dy - 1; ix = x + dx - 1;
                     valid = ((unsigned)iy < 28u) && ((unsigned)ix < 28u); }
    else { iy = y + dy; ix = x + dx; }
    const _Float16* bp = valid
        ? (XH + ((size_t)n * HW + iy * 28 + ix) * COUT + g * 32 + hs * 8)
        : (zp + hs * 8);
    v16h B = ldpair(bp);
    v16h A = ldpair(Wc + (size_t)((g * 9 + t) * 32 + mt * 16 + ln) * 32 + hs * 8);
    acc = wmma16(A, B, acc);
  }
#pragma unroll
  for (int j = 0; j < 8; ++j) {
    int ch = mt * 16 + j + 8 * hs;
    float v = acc[j];
    if (mode == 0) {
      Out[((size_t)(n * 8 + g) * HW + p) * CG + ch] = (_Float16)v;
    } else if (mode == 1) {
      if (p < 676) Out[((size_t)(n * 8 + g) * KD + p) * CG + ch] = (_Float16)v;
    } else {
      if (p < 676) Out[((size_t)(n * 8 + g) * CG + ch) * VROW + p] = (_Float16)v;
    }
  }
}

// ---------------- flash attention per (n, g, 16-query tile) ----------------
__global__ void __launch_bounds__(32) k_attn(const _Float16* __restrict__ Q,
                                             const _Float16* __restrict__ K,
                                             const _Float16* __restrict__ V,
                                             const float* __restrict__ tau,
                                             _Float16* __restrict__ Ab) {
  __shared__ _Float16 lds[16 * 40];  // P staging, 80B row stride
  int id = blockIdx.x;
  int qt = id % 49, g = (id / 49) & 7, n = id / 392;
  int l = threadIdx.x, hs = l >> 4, ln = l & 15;
  float tg = tau[g];
  v16h aQ = ldpair(Q + ((size_t)(n * 8 + g) * HW + qt * 16 + ln) * CG + hs * 8);
  size_t kb = (size_t)(n * 8 + g) * KD * CG;
  size_t vb = (size_t)(n * 8 + g) * CG * VROW;
  v8f acc0 = {0, 0, 0, 0, 0, 0, 0, 0}, acc1 = {0, 0, 0, 0, 0, 0, 0, 0};
  const v8f zc = {0, 0, 0, 0, 0, 0, 0, 0};
  float m[8], ls[8];
#pragma unroll
  for (int e = 0; e < 8; ++e) { m[e] = -1e30f; ls[e] = 0.0f; }

  for (int it = 0; it < 22; ++it) {
    int k0 = it * 32;
    float st0[8], st1[8];
    {
      int key = k0 + ln; int kc = key < 676 ? key : 675;
      v8f s = wmma16(aQ, ldpair(K + kb + (size_t)kc * CG + hs * 8), zc);
      bool ok = key < 676;
#pragma unroll
      for (int e = 0; e < 8; ++e) st0[e] = ok ? s[e] * tg : -1e30f;
    }
    {
      int key = k0 + 16 + ln; int kc = key < 676 ? key : 675;
      v8f s = wmma16(aQ, ldpair(K + kb + (size_t)kc * CG + hs * 8), zc);
      bool ok = key < 676;
#pragma unroll
      for (int e = 0; e < 8; ++e) st1[e] = ok ? s[e] * tg : -1e30f;
    }
#pragma unroll
    for (int e = 0; e < 8; ++e) {
      float mx = fmaxf(st0[e], st1[e]);
      mx = fmaxf(mx, __shfl_xor(mx, 1, 16));
      mx = fmaxf(mx, __shfl_xor(mx, 2, 16));
      mx = fmaxf(mx, __shfl_xor(mx, 4, 16));
      mx = fmaxf(mx, __shfl_xor(mx, 8, 16));
      float mn = fmaxf(m[e], mx);
      float al = __expf(m[e] - mn);
      m[e] = mn;
      float e0 = __expf(st0[e] - mn), e1 = __expf(st1[e] - mn);
      float rs = e0 + e1;
      rs += __shfl_xor(rs, 1, 16);
      rs += __shfl_xor(rs, 2, 16);
      rs += __shfl_xor(rs, 4, 16);
      rs += __shfl_xor(rs, 8, 16);
      ls[e] = ls[e] * al + rs;
      acc0[e] *= al; acc1[e] *= al;
      lds[(e + 8 * hs) * 40 + ln]      = (_Float16)e0;
      lds[(e + 8 * hs) * 40 + 16 + ln] = (_Float16)e1;
    }
    __syncthreads();                         // single-wave WG: compiler fence, hw S_NOP
    v16h aP = ldpair(&lds[ln * 40 + hs * 8]);
    __syncthreads();
    acc0 = wmma16(aP, ldpair(V + vb + (size_t)ln * VROW + k0 + hs * 8), acc0);
    acc1 = wmma16(aP, ldpair(V + vb + (size_t)(16 + ln) * VROW + k0 + hs * 8), acc1);
  }
#pragma unroll
  for (int e = 0; e < 8; ++e) {
    float r = 1.0f / ls[e];
    size_t ab = ((size_t)(n * 8 + g) * HW + qt * 16 + e + 8 * hs) * CG;
    Ab[ab + ln]      = (_Float16)(acc0[e] * r);
    Ab[ab + 16 + ln] = (_Float16)(acc1[e] * r);
  }
}

// ---------------- fused gates + LSTM epilogue ----------------
__global__ void __launch_bounds__(32) k_gates(
    const _Float16* __restrict__ XH, const _Float16* __restrict__ Ab,
    const _Float16* __restrict__ WIA, const _Float16* __restrict__ WGA,
    const _Float16* __restrict__ WOA, const _Float16* __restrict__ WIX,
    const _Float16* __restrict__ WGX, const _Float16* __restrict__ WOX,
    const float* __restrict__ bi, const float* __restrict__ bg,
    const float* __restrict__ bo, const _Float16* __restrict__ zp,
    float* __restrict__ out) {
  int id = blockIdx.x;
  int pt = id % 49, g = (id / 49) & 7, n = id / 392;
  int l = threadIdx.x, hs = l >> 4, ln = l & 15;
  int p = pt * 16 + ln;
  const _Float16* WA[3]  = {WIA, WGA, WOA};
  const _Float16* WXc[3] = {WIX, WGX, WOX};
  v8f acc[6];
  const v8f zc = {0, 0, 0, 0, 0, 0, 0, 0};
#pragma unroll
  for (int k = 0; k < 6; ++k) acc[k] = zc;

  v16h Ba = ldpair(Ab + ((size_t)(n * 8 + g) * HW + p) * CG + hs * 8);
#pragma unroll
  for (int gi = 0; gi < 3; ++gi)
#pragma unroll
    for (int hf = 0; hf < 2; ++hf) {
      v16h Aw = ldpair(WA[gi] + (size_t)(g * 32 + hf * 16 + ln) * 32 + hs * 8);
      acc[gi * 2 + hf] = wmma16(Aw, Ba, acc[gi * 2 + hf]);
    }
  int y = p / 28, x = p % 28;
#pragma unroll
  for (int t = 0; t < 9; ++t) {
    int dy = t / 3, dx = t % 3;
    int iy = y + dy - 1, ix = x + dx - 1;
    bool valid = ((unsigned)iy < 28u) && ((unsigned)ix < 28u);
    const _Float16* bp = valid
        ? (XH + ((size_t)n * HW + iy * 28 + ix) * COUT + g * 32 + hs * 8)
        : (zp + hs * 8);
    v16h B = ldpair(bp);
#pragma unroll
    for (int gi = 0; gi < 3; ++gi)
#pragma unroll
      for (int hf = 0; hf < 2; ++hf) {
        v16h Aw = ldpair(WXc[gi] + (size_t)((g * 9 + t) * 32 + hf * 16 + ln) * 32 + hs * 8);
        acc[gi * 2 + hf] = wmma16(Aw, B, acc[gi * 2 + hf]);
      }
  }
  float bv[3][2];
#pragma unroll
  for (int hf = 0; hf < 2; ++hf) {
    bv[0][hf] = bi[g * 32 + hf * 16 + ln];
    bv[1][hf] = bg[g * 32 + hf * 16 + ln];
    bv[2][hf] = bo[g * 32 + hf * 16 + ln];
  }
#pragma unroll
  for (int hf = 0; hf < 2; ++hf)
#pragma unroll
    for (int j = 0; j < 8; ++j) {
      int src = j + 8 * hs;
      float pi = acc[0 * 2 + hf][j] + __shfl(bv[0][hf], src, 16);
      float pg = acc[1 * 2 + hf][j] + __shfl(bv[1][hf], src, 16);
      float po = acc[2 * 2 + hf][j] + __shfl(bv[2][hf], src, 16);
      float cc = sigm(pi) * ftanh(pg);       // c = i*g  (f*c0 == 0)
      float hh = sigm(po) * ftanh(cc);
      int cg_ = g * 32 + hf * 16 + j + 8 * hs;
      out[((size_t)n * COUT + cg_) * HW + p] = hh;
    }
}

// ---------------- host ----------------
extern "C" void kernel_launch(void* const* d_in, const int* in_sizes, int n_in,
                              void* d_out, int out_size, void* d_ws, size_t ws_size,
                              hipStream_t stream) {
  (void)in_sizes; (void)n_in; (void)out_size; (void)ws_size;
  const float* x_in = (const float*)d_in[0];
  const float* w_x  = (const float*)d_in[1];
  const float* w_q  = (const float*)d_in[3];
  const float* w_k  = (const float*)d_in[4];
  const float* w_v  = (const float*)d_in[5];
  const float* tau  = (const float*)d_in[6];
  const float* w_ia = (const float*)d_in[7];
  const float* w_ix = (const float*)d_in[8];
  const float* b_i  = (const float*)d_in[9];
  const float* w_ga = (const float*)d_in[13];
  const float* w_gx = (const float*)d_in[14];
  const float* b_g  = (const float*)d_in[15];
  const float* w_oa = (const float*)d_in[16];
  const float* w_ox = (const float*)d_in[17];
  const float* b_o  = (const float*)d_in[18];

  char* w = (char*)d_ws;
  size_t off = 0;
  auto alloc = [&](size_t bytes) -> void* {
    void* p = w + off;
    off += (bytes + 255) & ~(size_t)255;
    return p;
  };
  _Float16* zp  = (_Float16*)alloc(1024);
  _Float16* WX  = (_Float16*)alloc((size_t)COUT * CIN * 2);
  _Float16* WQ  = (_Float16*)alloc((size_t)G * 9 * CG * CG * 2);
  _Float16* WK  = (_Float16*)alloc((size_t)G * 9 * CG * CG * 2);
  _Float16* WV  = (_Float16*)alloc((size_t)G * 9 * CG * CG * 2);
  _Float16* WIX = (_Float16*)alloc((size_t)G * 9 * CG * CG * 2);
  _Float16* WGX = (_Float16*)alloc((size_t)G * 9 * CG * CG * 2);
  _Float16* WOX = (_Float16*)alloc((size_t)G * 9 * CG * CG * 2);
  _Float16* WIA = (_Float16*)alloc((size_t)G * CG * CG * 2);
  _Float16* WGA = (_Float16*)alloc((size_t)G * CG * CG * 2);
  _Float16* WOA = (_Float16*)alloc((size_t)G * CG * CG * 2);
  _Float16* XT  = (_Float16*)alloc((size_t)NB * HW * CIN * 2);
  _Float16* XH  = (_Float16*)alloc((size_t)NB * HW * COUT * 2);
  _Float16* Qb  = (_Float16*)alloc((size_t)NB * G * HW * CG * 2);
  _Float16* Kb  = (_Float16*)alloc((size_t)NB * G * KD * CG * 2);
  _Float16* Vb  = (_Float16*)alloc((size_t)NB * G * CG * VROW * 2 + 1024);
  _Float16* Ab  = (_Float16*)alloc((size_t)NB * G * HW * CG * 2);

  const int wc = G * 9 * CG * CG;
  k_zero16<<<2, 256, 0, stream>>>(zp, 512);
  k_cvt<<<(COUT * CIN + 255) / 256, 256, 0, stream>>>(w_x, WX, COUT * CIN);
  k_cvt_wconv<<<(wc + 255) / 256, 256, 0, stream>>>(w_q, WQ);
  k_cvt_wconv<<<(wc + 255) / 256, 256, 0, stream>>>(w_k, WK);
  k_cvt_wconv<<<(wc + 255) / 256, 256, 0, stream>>>(w_v, WV);
  k_cvt_wconv<<<(wc + 255) / 256, 256, 0, stream>>>(w_ix, WIX);
  k_cvt_wconv<<<(wc + 255) / 256, 256, 0, stream>>>(w_gx, WGX);
  k_cvt_wconv<<<(wc + 255) / 256, 256, 0, stream>>>(w_ox, WOX);
  k_cvt<<<(G * CG * CG + 255) / 256, 256, 0, stream>>>(w_ia, WIA, G * CG * CG);
  k_cvt<<<(G * CG * CG + 255) / 256, 256, 0, stream>>>(w_ga, WGA, G * CG * CG);
  k_cvt<<<(G * CG * CG + 255) / 256, 256, 0, stream>>>(w_oa, WOA, G * CG * CG);
  k_xt<<<(NB * HW * CIN + 255) / 256, 256, 0, stream>>>(x_in, XT);

  k_proj<<<NB * 16 * 49, 32, 0, stream>>>(XT, WX, XH);
  k_conv3<<<NB * G * 2 * 49, 32, 0, stream>>>(XH, WQ, Qb, zp, 0);
  k_conv3<<<NB * G * 2 * 43, 32, 0, stream>>>(XH, WK, Kb, zp, 1);
  k_conv3<<<NB * G * 2 * 43, 32, 0, stream>>>(XH, WV, Vb, zp, 2);
  k_attn<<<NB * G * 49, 32, 0, stream>>>(Qb, Kb, Vb, tau, Ab);
  k_gates<<<NB * G * 49, 32, 0, stream>>>(XH, Ab, WIA, WGA, WOA, WIX, WGX, WOX,
                                          b_i, b_g, b_o, zp, (float*)d_out);
}
